// OptimizedAttention_83949430768206
// MI455X (gfx1250) — matrix operand order repaired and checked by reading, not compiled, not run on hardware
//
#include <hip/hip_runtime.h>
#include <hip/hip_bf16.h>

#define SEQ     2048
#define DMODEL  4096
#define NH      32
#define NKV     8
#define DK      128

typedef __bf16 bf16;
typedef __attribute__((ext_vector_type(16))) __bf16 bf16x16;
typedef __attribute__((ext_vector_type(8)))  __bf16 bf16x8;
typedef __attribute__((ext_vector_type(4)))  __bf16 bf16x4;
typedef __attribute__((ext_vector_type(8)))  float  f32x8;
typedef __attribute__((ext_vector_type(4)))  float  f32x4;
typedef __attribute__((ext_vector_type(4)))  int    v4i;

// ---------------------------------------------------------------------------
// gfx1250 async global->LDS copy (GLOBAL_LOAD_ASYNC_TO_LDS_B128, ASYNCcnt),
// guarded so the kernel still compiles if the toolchain lacks the builtins.
// Builtin takes typed addrspace pointers: (v4i AS1*, v4i AS3*, imm, imm).
// ---------------------------------------------------------------------------
#if __has_builtin(__builtin_amdgcn_global_load_async_to_lds_b128) && \
    __has_builtin(__builtin_amdgcn_s_wait_asynccnt)
#define ASYNC_LDS_COPY 1
#else
#define ASYNC_LDS_COPY 0
#endif

__device__ __forceinline__ void cp16_g2l(void* lds, const void* g) {
#if ASYNC_LDS_COPY
  __builtin_amdgcn_global_load_async_to_lds_b128(
      (__attribute__((address_space(1))) v4i*)g,
      (__attribute__((address_space(3))) v4i*)lds, 0, 0);
#else
  *(bf16x8*)lds = *(const bf16x8*)g;
#endif
}
__device__ __forceinline__ void cp_wait() {
#if ASYNC_LDS_COPY
  __builtin_amdgcn_s_wait_asynccnt(0);
#endif
}

// Build a 16-element bf16 WMMA fragment from two 16-byte LDS reads.
__device__ __forceinline__ bf16x16 ldfrag(const bf16* p0, const bf16* p1) {
  bf16x16 r;
  bf16x8 a = *(const bf16x8*)p0;
  bf16x8 b = *(const bf16x8*)p1;
#pragma unroll
  for (int i = 0; i < 8; ++i) { r[i] = a[i]; r[i + 8] = b[i]; }
  return r;
}

__device__ __forceinline__ f32x8 wmma_bf16(bf16x16 a, bf16x16 b, f32x8 c) {
  return __builtin_amdgcn_wmma_f32_16x16x32_bf16(false, a, false, b, (short)0, c,
                                                 false, false);
}

// ---------------------------------------------------------------------------
// C[M,N] = A[M,K] @ B[N,K]^T   (B row-major [N,K], i.e. x @ W.T)
// Block tile 128x64, BK=32. 8 waves arranged 4x2, each wave does 32x32.
// A is fp32 (converted to bf16 while staging) or already bf16 (async path).
// ---------------------------------------------------------------------------
template <bool A_BF16>
__global__ __launch_bounds__(256) void gemm_xwT(const void* __restrict__ Ap,
                                                const float* __restrict__ B,
                                                float* __restrict__ C,
                                                int M, int N, int K) {
  __shared__ bf16 sA[128 * 32];
  __shared__ bf16 sB[64 * 32];
  const int tid = threadIdx.x;
  const int wv = tid >> 5, lane = tid & 31, hi = lane >> 4, l15 = lane & 15;
  const int wm = (wv >> 1) * 32, wn = (wv & 1) * 32;
  const int m0 = blockIdx.y * 128, n0 = blockIdx.x * 64;

  const float* Af = (const float*)Ap;
  const bf16*  Ab = (const bf16*)Ap;
  f32x8 acc[2][2] = {};

  for (int kb = 0; kb < K; kb += 32) {
    __syncthreads();
    if (A_BF16) {
#pragma unroll
      for (int i = 0; i < 2; ++i) {                // 512 chunks of 8 bf16
        int cf = tid + i * 256;
        int row = cf >> 2, kq = cf & 3;
        cp16_g2l(sA + row * 32 + kq * 8,
                 Ab + (size_t)(m0 + row) * K + kb + kq * 8);
      }
    } else {
#pragma unroll
      for (int i = 0; i < 4; ++i) {                // 1024 chunks of float4
        int cf = tid + i * 256;
        int row = cf >> 3, kq = cf & 7;
        f32x4 v = *(const f32x4*)(Af + (size_t)(m0 + row) * K + kb + kq * 4);
        bf16x4 o;
        o[0] = (bf16)v[0]; o[1] = (bf16)v[1]; o[2] = (bf16)v[2]; o[3] = (bf16)v[3];
        *(bf16x4*)(sA + row * 32 + kq * 4) = o;
      }
    }
#pragma unroll
    for (int i = 0; i < 2; ++i) {                  // B tile: 64 rows x 32 K
      int cf = tid + i * 256;
      int row = cf >> 3, kq = cf & 7;
      f32x4 v = *(const f32x4*)(B + (size_t)(n0 + row) * K + kb + kq * 4);
      bf16x4 o;
      o[0] = (bf16)v[0]; o[1] = (bf16)v[1]; o[2] = (bf16)v[2]; o[3] = (bf16)v[3];
      *(bf16x4*)(sB + row * 32 + kq * 4) = o;
    }
    if (kb + 32 < K)  // gfx1250: lowers to global_prefetch_b8
      __builtin_prefetch(B + (size_t)(n0 + (tid >> 2)) * K + kb + 32, 0, 0);
    if (A_BF16) cp_wait();
    __syncthreads();

    bf16x16 afr[2], bfr[2];
#pragma unroll
    for (int mi = 0; mi < 2; ++mi) {
      const bf16* ba = sA + (wm + mi * 16 + l15) * 32 + hi * 8;
      afr[mi] = ldfrag(ba, ba + 16);               // A: K 0-7/8-15 then 16-23/24-31
    }
#pragma unroll
    for (int ni = 0; ni < 2; ++ni) {
      const bf16* bb = sB + (wn + ni * 16 + l15) * 32 + hi * 8;
      bfr[ni] = ldfrag(bb, bb + 16);               // B: K 0-7/8-15 then 16-23/24-31
    }
#pragma unroll
    for (int mi = 0; mi < 2; ++mi)
#pragma unroll
      for (int ni = 0; ni < 2; ++ni)
        acc[mi][ni] = wmma_bf16(afr[mi], bfr[ni], acc[mi][ni]);
  }

  // f32 C layout: VGPR r -> M = r + 8*hi, N = lane&15
#pragma unroll
  for (int mi = 0; mi < 2; ++mi)
#pragma unroll
    for (int ni = 0; ni < 2; ++ni)
#pragma unroll
      for (int r = 0; r < 8; ++r) {
        int row = m0 + wm + mi * 16 + r + hi * 8;
        int col = n0 + wn + ni * 16 + l15;
        C[(size_t)row * N + col] = acc[mi][ni][r];
      }
}

// ---------------------------------------------------------------------------
// Per-head LayerNorm over D_K=128 + bf16 pack. One wave per head-row.
// headMajor=0 -> dst[token][h*128+d]; headMajor=1 -> dst[h][token][d]
// ---------------------------------------------------------------------------
__global__ __launch_bounds__(256) void ln_pack(const float* __restrict__ src,
                                               bf16* __restrict__ dst, int H,
                                               const float* __restrict__ gamma,
                                               const float* __restrict__ beta,
                                               int headMajor) {
  const int wv = threadIdx.x >> 5, lane = threadIdx.x & 31;
  const int rowIdx = blockIdx.x * 8 + wv;  // token*H + h
  const int token = rowIdx / H, h = rowIdx % H;

  f32x4 v = *(const f32x4*)(src + (size_t)rowIdx * DK + lane * 4);
  float s  = v[0] + v[1] + v[2] + v[3];
  float sq = v[0] * v[0] + v[1] * v[1] + v[2] * v[2] + v[3] * v[3];
#pragma unroll
  for (int off = 16; off > 0; off >>= 1) {  // wave32 reduction
    s  += __shfl_xor(s, off, 32);
    sq += __shfl_xor(sq, off, 32);
  }
  float mean = s * (1.f / DK);
  float var  = sq * (1.f / DK) - mean * mean;
  float rstd = rsqrtf(var + 1e-6f);

  size_t o = headMajor ? ((size_t)h * SEQ + token) * DK + lane * 4
                       : (size_t)rowIdx * DK + lane * 4;
  bf16x4 ov;
#pragma unroll
  for (int i = 0; i < 4; ++i)
    ov[i] = (bf16)((v[i] - mean) * rstd * gamma[lane * 4 + i] + beta[lane * 4 + i]);
  *(bf16x4*)(dst + o) = ov;
}

// ---------------------------------------------------------------------------
// Pack V transposed: v_raw[token][h*128+d] -> Vt[h][d][token] (bf16)
// ---------------------------------------------------------------------------
__global__ __launch_bounds__(256) void v_pack_t(const float* __restrict__ v_raw,
                                                bf16* __restrict__ Vt) {
  int tid = blockIdx.x * 256 + threadIdx.x;  // 0 .. 8*128*2048-1
  int t  = tid & (SEQ - 1);
  int hd = tid >> 11;                        // h*128 + d
  Vt[(size_t)hd * SEQ + t] = (bf16)v_raw[(size_t)t * (NKV * DK) + hd];
}

// ---------------------------------------------------------------------------
// Flash attention, causal, GQA rep=4. Block = 1 head x 128 queries (8 waves,
// 16 query rows each). 64-key blocks; 32 bf16 WMMAs per block per wave.
// K/V/Q tiles staged via async global->LDS copies when available.
// ---------------------------------------------------------------------------
__global__ __launch_bounds__(256) void flash_attn(const bf16* __restrict__ Qb,
                                                  const bf16* __restrict__ Kb,
                                                  const bf16* __restrict__ Vt,
                                                  bf16* __restrict__ Ob) {
  __shared__ bf16 sK[64 * 128];    // [token][d]
  __shared__ bf16 sV[128 * 64];    // [d][token]
  __shared__ bf16 sQP[128 * 128];  // Q tile, then per-wave P regions
  const int tid = threadIdx.x, wv = tid >> 5, lane = tid & 31;
  const int hi = lane >> 4, l15 = lane & 15;
  const int q0 = blockIdx.x * 128, h = blockIdx.y, hkv = h >> 2;
  const float scale = 0.08838834764831845f;  // 1/sqrt(128)

  // Stage Q tile (128 x 128 bf16)
#pragma unroll
  for (int i = 0; i < 8; ++i) {
    int cf = tid + i * 256;
    int row = cf >> 4, c8 = cf & 15;
    cp16_g2l(sQP + row * 128 + c8 * 8,
             Qb + (size_t)(q0 + row) * DMODEL + h * DK + c8 * 8);
  }
  cp_wait();
  __syncthreads();
  bf16x16 qfr[4];
#pragma unroll
  for (int c = 0; c < 4; ++c) {
    const bf16* base = sQP + (wv * 16 + l15) * 128 + c * 32 + hi * 8;
    qfr[c] = ldfrag(base, base + 16);
  }
  __syncthreads();  // sQP now reusable as P

  f32x8 oacc[8] = {};
  float m_i[8], l_i[8];
#pragma unroll
  for (int r = 0; r < 8; ++r) { m_i[r] = -1e30f; l_i[r] = 0.f; }

  const int nkb = (q0 >> 6) + 2;  // causal: keys <= q0+127
  for (int ib = 0; ib < nkb; ++ib) {
    const int kb = ib * 64;
    __syncthreads();
#pragma unroll
    for (int i = 0; i < 4; ++i) {   // K tile 64x128
      int cf = tid + i * 256;
      int row = cf >> 4, c8 = cf & 15;
      cp16_g2l(sK + row * 128 + c8 * 8,
               Kb + ((size_t)hkv * SEQ + kb + row) * DK + c8 * 8);
    }
#pragma unroll
    for (int i = 0; i < 4; ++i) {   // V^T tile 128x64
      int cf = tid + i * 256;
      int row = cf >> 3, c8 = cf & 7;
      cp16_g2l(sV + row * 64 + c8 * 8,
               Vt + ((size_t)hkv * DK + row) * SEQ + kb + c8 * 8);
    }
    cp_wait();
    __syncthreads();

    // S = Q K^T  (16 x 64, fp32)
    f32x8 sacc[4] = {};
#pragma unroll
    for (int ni = 0; ni < 4; ++ni)
#pragma unroll
      for (int c = 0; c < 4; ++c) {
        const bf16* base = sK + (ni * 16 + l15) * 128 + c * 32 + hi * 8;
        sacc[ni] = wmma_bf16(qfr[c], ldfrag(base, base + 16), sacc[ni]);
      }

    // scale + causal mask + row max (rows live in lanes of same hi group)
    float cmax[8];
    int row_g[8];
#pragma unroll
    for (int r = 0; r < 8; ++r) { cmax[r] = -1e30f; row_g[r] = q0 + wv * 16 + r + hi * 8; }
#pragma unroll
    for (int ni = 0; ni < 4; ++ni) {
      int col = kb + ni * 16 + l15;
#pragma unroll
      for (int r = 0; r < 8; ++r) {
        float v = sacc[ni][r] * scale;
        v = (col > row_g[r]) ? -1e30f : v;
        sacc[ni][r] = v;
        cmax[r] = fmaxf(cmax[r], v);
      }
    }
#pragma unroll
    for (int r = 0; r < 8; ++r)
#pragma unroll
      for (int off = 8; off > 0; off >>= 1)
        cmax[r] = fmaxf(cmax[r], __shfl_xor(cmax[r], off, 32));

    float alpha[8], rs[8];
#pragma unroll
    for (int r = 0; r < 8; ++r) {
      float mn = fmaxf(m_i[r], cmax[r]);
      alpha[r] = __expf(m_i[r] - mn);
      m_i[r] = mn;
      rs[r] = 0.f;
    }

    // P = exp(S - m) -> bf16 into this wave's private LDS region
    bf16* P = sQP + wv * (16 * 64);
#pragma unroll
    for (int ni = 0; ni < 4; ++ni)
#pragma unroll
      for (int r = 0; r < 8; ++r) {
        float p = __expf(sacc[ni][r] - m_i[r]);
        rs[r] += p;
        P[(r + hi * 8) * 64 + ni * 16 + l15] = (bf16)p;
      }
#pragma unroll
    for (int r = 0; r < 8; ++r) {
#pragma unroll
      for (int off = 8; off > 0; off >>= 1)
        rs[r] += __shfl_xor(rs[r], off, 32);
      l_i[r] = l_i[r] * alpha[r] + rs[r];
    }
#pragma unroll
    for (int ni = 0; ni < 8; ++ni)
#pragma unroll
      for (int r = 0; r < 8; ++r)
        oacc[ni][r] *= alpha[r];

    // O += P V
    bf16x16 pfr[2];
#pragma unroll
    for (int c = 0; c < 2; ++c) {
      const bf16* base = P + l15 * 64 + c * 32 + hi * 8;
      pfr[c] = ldfrag(base, base + 16);
    }
#pragma unroll
    for (int ni = 0; ni < 8; ++ni)
#pragma unroll
      for (int c = 0; c < 2; ++c) {
        const bf16* base = sV + (ni * 16 + l15) * 64 + c * 32 + hi * 8;
        oacc[ni] = wmma_bf16(pfr[c], ldfrag(base, base + 16), oacc[ni]);
      }
  }

  // normalize + write bf16 attention output (token-major, [token][h*128+d])
#pragma unroll
  for (int r = 0; r < 8; ++r) {
    float inv = 1.f / l_i[r];
    int row = q0 + wv * 16 + r + hi * 8;
#pragma unroll
    for (int ni = 0; ni < 8; ++ni)
      Ob[(size_t)row * DMODEL + h * DK + ni * 16 + l15] = (bf16)(oacc[ni][r] * inv);
  }
}

// ---------------------------------------------------------------------------
extern "C" void kernel_launch(void* const* d_in, const int* in_sizes, int n_in,
                              void* d_out, int out_size, void* d_ws, size_t ws_size,
                              hipStream_t stream) {
  (void)in_sizes; (void)n_in; (void)out_size; (void)ws_size;
  const float* x   = (const float*)d_in[0];
  const float* Wq  = (const float*)d_in[3];
  const float* Wk  = (const float*)d_in[4];
  const float* Wv  = (const float*)d_in[5];
  const float* Wo  = (const float*)d_in[6];
  const float* qg  = (const float*)d_in[7];
  const float* qbt = (const float*)d_in[8];
  const float* kg  = (const float*)d_in[9];
  const float* kbt = (const float*)d_in[10];

  char* ws = (char*)d_ws;
  float* q_raw = (float*)ws;                          // 32 MB
  float* k_raw = (float*)(ws + ((size_t)32 << 20));   //  8 MB
  float* v_raw = (float*)(ws + ((size_t)40 << 20));   //  8 MB
  bf16*  Qbuf  = (bf16*)(ws + ((size_t)48 << 20));    // 16 MB
  bf16*  Kbuf  = (bf16*)(ws + ((size_t)64 << 20));    //  4 MB
  bf16*  Vt    = (bf16*)(ws + ((size_t)68 << 20));    //  4 MB
  bf16*  attn  = (bf16*)q_raw;  // alias: q_raw dead after ln_pack(q)

  // QKV projections (fp32 -> bf16 WMMA, fp32 out)
  gemm_xwT<false><<<dim3(DMODEL / 64, SEQ / 128), 256, 0, stream>>>(x, Wq, q_raw, SEQ, DMODEL, DMODEL);
  gemm_xwT<false><<<dim3((NKV * DK) / 64, SEQ / 128), 256, 0, stream>>>(x, Wk, k_raw, SEQ, NKV * DK, DMODEL);
  gemm_xwT<false><<<dim3((NKV * DK) / 64, SEQ / 128), 256, 0, stream>>>(x, Wv, v_raw, SEQ, NKV * DK, DMODEL);

  // LayerNorm + bf16 pack
  ln_pack<<<SEQ * NH / 8, 256, 0, stream>>>(q_raw, Qbuf, NH, qg, qbt, 0);
  ln_pack<<<SEQ * NKV / 8, 256, 0, stream>>>(k_raw, Kbuf, NKV, kg, kbt, 1);
  v_pack_t<<<(SEQ * NKV * DK) / 256, 256, 0, stream>>>(v_raw, Vt);

  // Causal GQA flash attention
  flash_attn<<<dim3(SEQ / 128, NH), 256, 0, stream>>>(Qbuf, Kbuf, Vt, attn);

  // Output projection (bf16 A straight from workspace)
  gemm_xwT<true><<<dim3(DMODEL / 64, SEQ / 128), 256, 0, stream>>>(attn, Wo, (float*)d_out, SEQ, DMODEL, DMODEL);
}
